// CoordinatedDREAMStack_87110526698094
// MI455X (gfx1250) — matrix operand order repaired and checked by reading, not compiled, not run on hardware
//
#include <hip/hip_runtime.h>

typedef _Float16 half_t;
typedef __attribute__((ext_vector_type(16))) _Float16 v16h;
typedef __attribute__((ext_vector_type(8)))  _Float16 v8h;
typedef __attribute__((ext_vector_type(8)))  float    v8f;

#define TPB    256
#define NWG    32
#define T_STEPS 128
#define BATCH  512
#define HIDDEN 256
#define RANK_R 32

struct DreamParams {
  const float* x;
  const float* eta[3];
  const float* tauv[3];
  const float* m1b[3];
  const float* m2b[3];
  const float* outb;
  const half_t* wC[3];
  const half_t* wB[3];
  const half_t* wV[3];
  const half_t* wW[3];
  const half_t* wM1[3];
  const half_t* wM2[3];
  const half_t* wOut;
  float* U;        // [3][BATCH][HIDDEN][RANK_R]
  float* em[3];
  float* ev[3];
  float* modp;     // [(T+1)][2][NWG]
  int*   counters; // [T]
  float* out;      // [BATCH][T][80]
};

// Float arrays first (16B-aligned offsets), then f16 shadows, then scalars.
struct SMem {
  float H[3][16*HIDDEN];    // recurrent hidden states f32 (persist across t)
  float Xin[16*96];         // padded layer-0 input f32
  float P[16*HIDDEN];       // scratch: x_pred / fast_raw / modulation / out
  float Q[16*HIDDEN];       // x_in @ B
  float EV[16*RANK_R];
  float VX[16*RANK_R];
  float G[16*64];           // m1 gemm output (pre-relu)
  float Red[16*TPB];        // reduction scratch; ALIASED as err@W output (R)
  half_t H16[3][16*HIDDEN]; // f16 shadows of H (gemm A operands)
  half_t Xin16[16*96];
  half_t Err16[16*HIDDEN];  // f16 error (padded cols zero)
  half_t G16[16*64];        // relu(m1) f16
  float surp[16];
  float tmp16[16];
  float usc[3][16];         // U norm-clip scale (persists across t)
  float modmean;
};

// 16xN = A(16xK, f16 row-major in LDS) @ preswizzled f16 weight frags, f32 accum.
// A lane layout (16-bit A 16x32): m=lane&15, hi=lane>>4, k = hi*8 + j + (j>=8?8:0)
//   -> per lane: two contiguous 8-half chunks at [kt*32+hi*8] and [kt*32+hi*8+16].
// B frag: one contiguous 32B v16h per lane. D: n=lane&15, row=hi*8+v.
// Shapes are compile-time: NNT>=16 runs two unconditional N-tile accumulators
// per K-loop (A loaded once for nt=wave and nt=wave+8, no EXEC juggling);
// NNT<=8 has a single wave-level guard per call.
template <int NKT, int NNT>
__device__ __forceinline__ void gemm16(const half_t* __restrict__ A, int lda,
                                       const half_t* __restrict__ WF,
                                       float* __restrict__ O, int ldo,
                                       int wave, int lane) {
  static_assert(NNT <= 16, "one or two N-tiles per wave");
  const int m  = lane & 15;
  const int hi = lane >> 4;
  const half_t* arow = A + m*lda + hi*8;
  if constexpr (NNT == 16) {
    const int nt0 = wave, nt1 = wave + 8;
    v8f acc0 = {}, acc1 = {};
#pragma unroll
    for (int kt = 0; kt < NKT; ++kt) {
      const v8h alo = *(const v8h*)(arow + kt*32);
      const v8h ahi = *(const v8h*)(arow + kt*32 + 16);
      const v16h a = __builtin_shufflevector(alo, ahi,
                       0,1,2,3,4,5,6,7,8,9,10,11,12,13,14,15);
      const v16h b0 = *(const v16h*)(WF + (((size_t)kt*NNT + nt0)*32 + lane)*16);
      const v16h b1 = *(const v16h*)(WF + (((size_t)kt*NNT + nt1)*32 + lane)*16);
      acc0 = __builtin_amdgcn_wmma_f32_16x16x32_f16(false, a, false, b0,
                                                    (short)0, acc0, false, false);
      acc1 = __builtin_amdgcn_wmma_f32_16x16x32_f16(false, a, false, b1,
                                                    (short)0, acc1, false, false);
    }
    const int n0 = nt0*16 + m;
    const int n1 = nt1*16 + m;
#pragma unroll
    for (int v = 0; v < 8; ++v) O[(hi*8 + v)*ldo + n0] = acc0[v];
#pragma unroll
    for (int v = 0; v < 8; ++v) O[(hi*8 + v)*ldo + n1] = acc1[v];
  } else {
    if (wave < NNT) {   // wave-uniform guard, once per call
      const int nt0 = wave;
      v8f acc0 = {};
#pragma unroll
      for (int kt = 0; kt < NKT; ++kt) {
        const v8h alo = *(const v8h*)(arow + kt*32);
        const v8h ahi = *(const v8h*)(arow + kt*32 + 16);
        const v16h a = __builtin_shufflevector(alo, ahi,
                         0,1,2,3,4,5,6,7,8,9,10,11,12,13,14,15);
        const v16h b0 = *(const v16h*)(WF + (((size_t)kt*NNT + nt0)*32 + lane)*16);
        acc0 = __builtin_amdgcn_wmma_f32_16x16x32_f16(false, a, false, b0,
                                                      (short)0, acc0, false, false);
      }
      const int n0 = nt0*16 + m;
#pragma unroll
      for (int v = 0; v < 8; ++v) O[(hi*8 + v)*ldo + n0] = acc0[v];
    }
  }
}

// Two-stage reduction of Red[16][TPB] row partials -> raw row sums in out16.
__device__ __forceinline__ void reduce_rows16(float* __restrict__ Red,
                                              float* __restrict__ out16, int tid) {
  __syncthreads();
  const int r = tid >> 4, c = tid & 15;
  float s = 0.f;
#pragma unroll
  for (int k = 0; k < 16; ++k) s += Red[r*TPB + c + k*16];
  Red[r*TPB + c] = s;       // slot (r,c) only re-read by its own writer (k=0)
  __syncthreads();
  if (tid < 16) {
    float t2 = 0.f;
#pragma unroll
    for (int k = 0; k < 16; ++k) t2 += Red[tid*TPB + k];
    out16[tid] = t2;
  }
  __syncthreads();
}

__device__ __forceinline__ void grid_barrier(int* cnt, int tid) {
  __syncthreads();
  if (tid == 0) {
    __threadfence();
    atomicAdd(cnt, 1);
    while (atomicAdd(cnt, 0) < NWG) __builtin_amdgcn_s_sleep(1);
    __threadfence();
  }
  __syncthreads();
}

__global__ __launch_bounds__(TPB, 1)
void dream_stack_kernel(DreamParams P) {
  __shared__ SMem sm;
  const int tid  = threadIdx.x;
  const int wave = tid >> 5;
  const int lane = tid & 31;
  const int wg   = blockIdx.x;
  const int bg0  = wg * 16;

  for (int i = tid; i < 3*16*HIDDEN; i += TPB) {
    (&sm.H[0][0])[i] = 0.f;
    (&sm.H16[0][0])[i] = (half_t)0.f;
  }
  if (tid < 16) { sm.usc[0][tid] = 1.f; sm.usc[1][tid] = 1.f; sm.usc[2][tid] = 1.f; }
  __syncthreads();

  for (int t = 0; t < T_STEPS; ++t) {
    // stage layer-0 input (pad 80 -> 96 with zeros), f32 + f16 shadow
    for (int i = tid; i < 16*96; i += TPB) {
      const int b = i / 96, d = i - b*96;
      const float v = (d < 80) ? P.x[((size_t)(bg0 + b)*T_STEPS + t)*80 + d] : 0.f;
      sm.Xin[i]   = v;
      sm.Xin16[i] = (half_t)v;
    }
    __syncthreads();

#pragma unroll
    for (int L = 0; L < 3; ++L) {
      constexpr int DIN0 = 80, DINP0 = 96;
      const int din  = (L == 0) ? DIN0  : HIDDEN;
      const int dinp = (L == 0) ? DINP0 : HIDDEN;
      const int nktK = (L == 0) ? DINP0/32 : HIDDEN/32;   // K-tiles over input dim
      (void)nktK;
      const float*  xin   = (L == 0) ? sm.Xin   : sm.H[L-1];
      const half_t* xin16 = (L == 0) ? sm.Xin16 : sm.H16[L-1];
      float* emG = P.em[L] + (size_t)bg0 * dinp;
      float* evG = P.ev[L] + (size_t)bg0 * dinp;

      // batch-wide modulation mean from previous step (deterministic 32-slot sum)
      if (tid == 0) {
        float s = 0.5f; // -> etaScale == 1.0 for top layer
        if (L < 2) {
          const float* mp = P.modp + ((size_t)t*2 + L) * NWG;
          float acc = 0.f;
          for (int w = 0; w < NWG; ++w) acc += mp[w];
          s = acc * (1.f / (512.f * 256.f));
        }
        sm.modmean = s;
      }

      // x_pred_pre = h @ C
      if constexpr (true) {
        if (L == 0) gemm16<8, 5>(sm.H16[L], HIDDEN, P.wC[L], sm.P, DINP0, wave, lane);
        else        gemm16<8,16>(sm.H16[L], HIDDEN, P.wC[L], sm.P, HIDDEN, wave, lane);
      }
      __syncthreads();

      // error, running stats, surprise partials
      for (int b = 0; b < 16; ++b) {
        float zpb = 0.f;
        const int d = tid;
        if (d < dinp) {
          float e = 0.f;
          if (d < din) {
            const int ii = b*dinp + d;
            const float xp = tanhf(sm.P[ii]);
            e = xin[ii] - xp;
            float em_o, ev_o;
            if (t == 0) { em_o = 0.f; ev_o = 1.f; }
            else        { em_o = emG[ii]; ev_o = evG[ii]; }
            const float z = (e - em_o) * rsqrtf(ev_o + 1e-6f);
            zpb = z * z;
            const float em_n = 0.9f*em_o + 0.1f*e;
            const float dv = e - em_n;
            emG[ii] = em_n;
            evG[ii] = 0.9f*ev_o + 0.1f*dv*dv;
          }
          sm.Err16[b*dinp + d] = (half_t)e;
        }
        sm.Red[b*TPB + tid] = zpb;
      }
      reduce_rows16(sm.Red, sm.surp, tid);
      if (tid < 16) {
        const float sur = sqrtf(sm.surp[tid] / (float)din);
        sm.surp[tid] = fminf(fmaxf(sur, 0.f), 10.f);
      }
      __syncthreads();

      // eV = err @ V ; Vx = x_in @ V
      if (L == 0) {
        gemm16<3, 2>(sm.Err16, DINP0, P.wV[L], sm.EV, RANK_R, wave, lane);
        gemm16<3, 2>(xin16,    DINP0, P.wV[L], sm.VX, RANK_R, wave, lane);
      } else {
        gemm16<8, 2>(sm.Err16, HIDDEN, P.wV[L], sm.EV, RANK_R, wave, lane);
        gemm16<8, 2>(xin16,    HIDDEN, P.wV[L], sm.VX, RANK_R, wave, lane);
      }
      __syncthreads();

      // Hebbian fast-weight pass over U (thread = h, loop b, vec over r)
      {
        const float etaScale = 1.f + 0.2f*(sm.modmean - 0.5f);
        const float etav = P.eta[L][tid] * etaScale;
        float4* Ub = (float4*)(P.U + (size_t)(L*BATCH + bg0) * HIDDEN * RANK_R);
        for (int b = 0; b < 16; ++b) {
          const float scp = sm.usc[L][b];
          const float hob = sm.H[L][b*HIDDEN + tid];
          const float ph  = etav * sm.surp[b] * hob;
          float ssq = 0.f, fr = 0.f;
          float4* up = Ub + ((size_t)b*HIDDEN + tid) * (RANK_R/4);
#pragma unroll
          for (int rc = 0; rc < RANK_R/4; ++rc) {
            float4 u;
            if (t == 0) { u.x = u.y = u.z = u.w = 0.f; }
            else { u = up[rc]; u.x *= scp; u.y *= scp; u.z *= scp; u.w *= scp; }
            const float4 e4 = *(const float4*)&sm.EV[b*RANK_R + rc*4];
            const float4 v4 = *(const float4*)&sm.VX[b*RANK_R + rc*4];
            float4 un;
            un.x = 0.99f*u.x + 0.1f*ph*e4.x;  // U + DT*(-FORGET*U + plast*heb)
            un.y = 0.99f*u.y + 0.1f*ph*e4.y;
            un.z = 0.99f*u.z + 0.1f*ph*e4.z;
            un.w = 0.99f*u.w + 0.1f*ph*e4.w;
            up[rc] = un;
            ssq += un.x*un.x + un.y*un.y + un.z*un.z + un.w*un.w;
            fr  += un.x*v4.x + un.y*v4.y + un.z*v4.z + un.w*v4.w;
          }
          sm.Red[b*TPB + tid] = ssq;
          sm.P[b*HIDDEN + tid] = fr;          // fast_raw
        }
        reduce_rows16(sm.Red, sm.tmp16, tid);
        if (tid < 16)
          sm.usc[L][tid] = fminf(1.0f / (sqrtf(sm.tmp16[tid]) + 1e-6f), 2.0f);
        __syncthreads();
      }

      // base drive and error injection (err@W output aliases Red scratch)
      if (L == 0) {
        gemm16<3,16>(xin16,    DINP0, P.wB[L], sm.Q,   HIDDEN, wave, lane);
        gemm16<3,16>(sm.Err16, DINP0, P.wW[L], sm.Red, HIDDEN, wave, lane);
      } else {
        gemm16<8,16>(xin16,    HIDDEN, P.wB[L], sm.Q,   HIDDEN, wave, lane);
        gemm16<8,16>(sm.Err16, HIDDEN, P.wW[L], sm.Red, HIDDEN, wave, lane);
      }
      __syncthreads();

      // LTC integration
      {
        const float tauSys = P.tauv[L][0];
        const float tl = 1.f + 0.5f * (float)L;
        for (int b = 0; b < 16; ++b) {
          const float sur  = sm.surp[b];
          const float ueff = (1.f + 0.5f*sur) * sm.Q[b*HIDDEN + tid]
                           + 0.1f * sm.usc[L][b] * sm.P[b*HIDDEN + tid];
          float tau = tauSys * tl / (1.f + sur);
          tau = fminf(fmaxf(tau, 0.01f), 50.f);
          float dtt = 0.1f / (tau + 0.1f);
          dtt = fminf(fmaxf(dtt, 0.01f), 0.5f);
          const float ho = sm.H[L][b*HIDDEN + tid];
          const float hl = (1.f - dtt)*ho + dtt*tanhf(ueff);
          const float hn = (hl + sm.Red[b*HIDDEN + tid]) * 0.99f + 0.01f*ho;
          sm.H[L][b*HIDDEN + tid]   = hn;
          sm.H16[L][b*HIDDEN + tid] = (half_t)hn;
        }
      }
      __syncthreads();

      // modulation head (only mods[1], mods[2] are ever consumed)
      if (L >= 1) {
        gemm16<8, 4>(sm.H16[L], HIDDEN, P.wM1[L], sm.G, 64, wave, lane);
        __syncthreads();
        for (int i = tid; i < 16*64; i += TPB) {
          const int d = i & 63;
          sm.G16[i] = (half_t)fmaxf(sm.G[i] + P.m1b[L][d], 0.f);
        }
        __syncthreads();
        gemm16<2,16>(sm.G16, 64, P.wM2[L], sm.P, HIDDEN, wave, lane);
        __syncthreads();
        float msum = 0.f;
        for (int b = 0; b < 16; ++b) {
          const float v = sm.P[b*HIDDEN + tid] + P.m2b[L][tid];
          msum += 1.f / (1.f + expf(-v));
        }
        sm.Red[tid] = msum;
        __syncthreads();
        if (tid < 16) {
          float s = 0.f;
#pragma unroll
          for (int k = 0; k < 16; ++k) s += sm.Red[tid + k*16];
          sm.Red[tid] = s;
        }
        __syncthreads();
        if (tid == 0) {
          float s = 0.f;
#pragma unroll
          for (int k = 0; k < 16; ++k) s += sm.Red[k];
          P.modp[((size_t)(t+1)*2 + (L-1)) * NWG + wg] = s;
        }
        __syncthreads();
      }
    } // layers

    // out_t = h3 @ out_w + out_b
    gemm16<8, 5>(sm.H16[2], HIDDEN, P.wOut, sm.P, 96, wave, lane);
    __syncthreads();
    for (int i = tid; i < 16*80; i += TPB) {
      const int b = i / 80, d = i - b*80;
      P.out[((size_t)(bg0 + b)*T_STEPS + t)*80 + d] = sm.P[b*96 + d] + P.outb[d];
    }

    grid_barrier(P.counters + t, tid);
  }
}

// Convert f32 (K,N) row-major weight to f16 fragments in WMMA B-matrix lane order:
// element i = ((kt*nNT+nt)*32 + lane)*16 + j  maps to  k=kt*32+(lane>>4)*8+j+(j>=8?8:0), n=nt*16+(lane&15)
__global__ void wconvert_kernel(const float* __restrict__ src, half_t* __restrict__ dst,
                                int Korig, int Kpad, int N) {
  const int total = Kpad * N;
  const int nNT = N >> 4;
  for (int i = blockIdx.x*blockDim.x + threadIdx.x; i < total; i += gridDim.x*blockDim.x) {
    const int j    = i & 15;
    const int lane = (i >> 4) & 31;
    const int fi   = i >> 9;
    const int kt   = fi / nNT;
    const int nt   = fi - kt*nNT;
    const int k = kt*32 + ((lane >> 4) << 3) + j + ((j >= 8) ? 8 : 0);
    const int n = nt*16 + (lane & 15);
    dst[i] = (half_t)((k < Korig) ? src[(size_t)k*N + n] : 0.f);
  }
}

extern "C" void kernel_launch(void* const* d_in, const int* in_sizes, int n_in,
                              void* d_out, int out_size, void* d_ws, size_t ws_size,
                              hipStream_t stream) {
  (void)in_sizes; (void)n_in; (void)out_size; (void)ws_size;
  char* base = (char*)d_ws;
  size_t off = 0;
  auto take = [&](size_t bytes) { size_t o = off; off += (bytes + 255) & ~(size_t)255; return o; };

  const size_t oCnt  = take(T_STEPS * sizeof(int));
  const size_t oModp = take((size_t)(T_STEPS + 1) * 2 * NWG * sizeof(float));
  size_t oEm[3], oEv[3];
  const int dinpA[3] = {96, 256, 256};
  for (int L = 0; L < 3; ++L) oEm[L] = take((size_t)BATCH * dinpA[L] * sizeof(float));
  for (int L = 0; L < 3; ++L) oEv[L] = take((size_t)BATCH * dinpA[L] * sizeof(float));
  const size_t oU = take((size_t)3 * BATCH * HIDDEN * RANK_R * sizeof(float));

  auto takeH = [&](size_t halfs) { return take(halfs * sizeof(half_t)); };
  size_t oC[3], oB[3], oV[3], oW[3], oM1[3], oM2[3];
  for (int L = 0; L < 3; ++L) {
    const int din = L ? 256 : 80, dinp = dinpA[L];
    oC[L]  = takeH((size_t)HIDDEN * din);
    oB[L]  = takeH((size_t)dinp * HIDDEN);
    oV[L]  = takeH((size_t)dinp * RANK_R);
    oW[L]  = takeH((size_t)dinp * HIDDEN);
    oM1[L] = takeH((size_t)HIDDEN * 64);
    oM2[L] = takeH((size_t)64 * HIDDEN);
  }
  const size_t oOut = takeH((size_t)HIDDEN * 80);

  // zero step counters + modp slot 0 (both live in first 2KB)
  hipMemsetAsync(d_ws, 0, 2048, stream);

  // d_in layout: [0]=x, then per layer (insertion order):
  // 0=C 1=B 2=V 3=W 4=eta 5=tau_sys 6=p1w 7=p1b 8=lng 9=lnb 10=p2w 11=p2b 12=m1w 13=m1b 14=m2w 15=m2b
  // then [49]=out_w, [50]=out_b
  auto LIN = [&](int L, int j) { return (const float*)d_in[1 + L*16 + j]; };
  auto conv = [&](const float* src, size_t dstOff, int Korig, int Kpad, int N) {
    const int total = Kpad * N;
    wconvert_kernel<<<(total + TPB - 1)/TPB, TPB, 0, stream>>>(
        src, (half_t*)(base + dstOff), Korig, Kpad, N);
  };
  for (int L = 0; L < 3; ++L) {
    const int din = L ? 256 : 80, dinp = dinpA[L];
    conv(LIN(L,0),  oC[L],  HIDDEN, HIDDEN, din);
    conv(LIN(L,1),  oB[L],  din, dinp, HIDDEN);
    conv(LIN(L,2),  oV[L],  din, dinp, RANK_R);
    conv(LIN(L,3),  oW[L],  din, dinp, HIDDEN);
    conv(LIN(L,12), oM1[L], HIDDEN, HIDDEN, 64);
    conv(LIN(L,14), oM2[L], 64, 64, HIDDEN);
  }
  conv((const float*)d_in[49], oOut, HIDDEN, HIDDEN, 80);

  DreamParams P;
  P.x = (const float*)d_in[0];
  for (int L = 0; L < 3; ++L) {
    P.eta[L]  = LIN(L, 4);
    P.tauv[L] = LIN(L, 5);
    P.m1b[L]  = LIN(L, 13);
    P.m2b[L]  = LIN(L, 15);
    P.wC[L]  = (const half_t*)(base + oC[L]);
    P.wB[L]  = (const half_t*)(base + oB[L]);
    P.wV[L]  = (const half_t*)(base + oV[L]);
    P.wW[L]  = (const half_t*)(base + oW[L]);
    P.wM1[L] = (const half_t*)(base + oM1[L]);
    P.wM2[L] = (const half_t*)(base + oM2[L]);
    P.em[L]  = (float*)(base + oEm[L]);
    P.ev[L]  = (float*)(base + oEv[L]);
  }
  P.outb     = (const float*)d_in[50];
  P.wOut     = (const half_t*)(base + oOut);
  P.U        = (float*)(base + oU);
  P.modp     = (float*)(base + oModp);
  P.counters = (int*)(base + oCnt);
  P.out      = (float*)d_out;

  dream_stack_kernel<<<NWG, TPB, 0, stream>>>(P);
}
